// SimpleLSTM_3977139716148
// MI455X (gfx1250) — compile-verified, matrix operand order checked by reading, not run on hardware
//
#include <hip/hip_runtime.h>
#include <hip/hip_bf16.h>
#include <math.h>

// LSTM LM forward for MI455X (gfx1250, wave32, WMMA bf16 16x16x32).
// VOCAB=32000, E=256, H=512, B=16, T=512.

typedef __attribute__((ext_vector_type(16))) __bf16 v16bf;
typedef __attribute__((ext_vector_type(8)))  float  v8f;

#define VOCAB 32000
#define EDIM  256
#define HDIM  512
#define BDIM  16
#define TDIM  512
#define G4H   (4 * HDIM)       // 2048 concatenated gate columns
#define ROWS  (BDIM * TDIM)    // 8192 (b*T + t) rows

__device__ __forceinline__ __bf16 f2bf(float f) {
  union { float f; unsigned u; } v; v.f = f;
  unsigned r = v.u + 0x7FFFu + ((v.u >> 16) & 1u);   // round-to-nearest-even
  unsigned short h = (unsigned short)(r >> 16);
  __bf16 o; __builtin_memcpy(&o, &h, 2); return o;
}

__device__ __forceinline__ float sigmoidf_(float x) { return 1.0f / (1.0f + __expf(-x)); }

// ---------------------------------------------------------------------------
// Prep: convert 4 gate weight matrices [K,H] f32 into concatenated transposed
// bf16 layout WT[(g*H + j)*K + k] = Wg[k*H + j]  (N-major, K-contiguous).
__global__ void prep_wT(const float* __restrict__ W0, const float* __restrict__ W1,
                        const float* __restrict__ W2, const float* __restrict__ W3,
                        __bf16* __restrict__ WT, int Kdim) {
  int idx = blockIdx.x * blockDim.x + threadIdx.x;
  int total = G4H * Kdim;
  if (idx >= total) return;
  int n = idx / Kdim, k = idx - n * Kdim;
  int g = n / HDIM, j = n - g * HDIM;
  const float* W = (g == 0) ? W0 : (g == 1) ? W1 : (g == 2) ? W2 : W3;
  WT[idx] = f2bf(W[(size_t)k * HDIM + j]);
}

// fcT[v*H + h] = fc_W[h*VOCAB + v]  (bf16, K-contiguous)
__global__ void prep_fcT(const float* __restrict__ fcW, __bf16* __restrict__ fcT) {
  int idx = blockIdx.x * blockDim.x + threadIdx.x;
  if (idx >= VOCAB * HDIM) return;
  int v = idx / HDIM, h = idx - v * HDIM;
  fcT[idx] = f2bf(fcW[(size_t)h * VOCAB + v]);
}

__global__ void prep_bcat(const float* __restrict__ b0, const float* __restrict__ b1,
                          const float* __restrict__ b2, const float* __restrict__ b3,
                          float* __restrict__ bc) {
  int i = blockIdx.x * blockDim.x + threadIdx.x;
  if (i >= G4H) return;
  int g = i / HDIM, j = i - g * HDIM;
  bc[i] = (g == 0 ? b0 : (g == 1 ? b1 : (g == 2 ? b2 : b3)))[j];
}

// Embedding gather -> bf16 activation rows X[(b*T+t)*E + e]
__global__ void gather_x(const int* __restrict__ tokens, const float* __restrict__ emb,
                         __bf16* __restrict__ X) {
  int idx = blockIdx.x * blockDim.x + threadIdx.x;
  if (idx >= ROWS * EDIM) return;
  int r = idx / EDIM, e = idx - r * EDIM;
  X[idx] = f2bf(emb[(size_t)tokens[r] * EDIM + e]);
}

// ---------------------------------------------------------------------------
// Register-blocked WMMA GEMM: C[M,N] = A[M,K](bf16 row-major) * BT[N,K](bf16) + bias[N].
// Each wave computes a 2x4 grid of 16x16 tiles (32 rows x 64 cols):
// per K-chunk -> 6 fragment loads feed 8 WMMAs (1.5 b128-pairs per WMMA),
// ~2.7x less L2->WGP traffic than the naive 1-tile-per-wave version.
// Block = 256 threads = 8 waves; wave w -> row-patch blockIdx.y*8+w, col-patch blockIdx.x.
// Requires M % 256 == 0, N % 64 == 0, K % 32 == 0 (true for all our shapes).
__global__ void wmma_gemm_bias(const __bf16* __restrict__ A, const __bf16* __restrict__ BT,
                               const float* __restrict__ bias, float* __restrict__ C,
                               int N, int K) {
  int wave = threadIdx.x >> 5;
  int lane = threadIdx.x & 31;
  int rc = lane & 15;              // A-row / BT-row (== B column) within a tile
  int kb = (lane >> 4) << 4;       // 0 or 16: which K-half this lane holds
  size_t m0 = ((size_t)blockIdx.y * 8 + wave) * 32;   // 32 output rows
  size_t n0 = (size_t)blockIdx.x * 64;                // 64 output cols

  const __bf16* ap[2];
  const __bf16* bp[4];
  #pragma unroll
  for (int mi = 0; mi < 2; ++mi) ap[mi] = A  + (m0 + mi * 16 + rc) * K + kb;
  #pragma unroll
  for (int ni = 0; ni < 4; ++ni) bp[ni] = BT + (n0 + ni * 16 + rc) * K + kb;

  v8f acc[2][4] = {};
  for (int k = 0; k < K; k += 32) {
    v16bf av[2], bv[4];
    #pragma unroll
    for (int mi = 0; mi < 2; ++mi) av[mi] = *(const v16bf*)(ap[mi] + k);
    #pragma unroll
    for (int ni = 0; ni < 4; ++ni) bv[ni] = *(const v16bf*)(bp[ni] + k);
    #pragma unroll
    for (int mi = 0; mi < 2; ++mi)
      #pragma unroll
      for (int ni = 0; ni < 4; ++ni)
        acc[mi][ni] = __builtin_amdgcn_wmma_f32_16x16x32_bf16(
            false, av[mi], false, bv[ni], (short)0, acc[mi][ni], false, false);
  }

  #pragma unroll
  for (int ni = 0; ni < 4; ++ni) {
    int n  = (int)n0 + ni * 16 + rc;
    float bb = bias[n];
    #pragma unroll
    for (int mi = 0; mi < 2; ++mi) {
      size_t mrow = m0 + mi * 16 + ((lane >> 4) << 3);  // D tile: vgpr g -> M = mrow+g
      #pragma unroll
      for (int g = 0; g < 8; ++g)
        C[(mrow + g) * N + n] = acc[mi][ni][g] + bb;
    }
  }
}

// ---------------------------------------------------------------------------
// One LSTM timestep. Grid = 32 blocks (16 H-columns each), block = 128 thr (4 waves).
// Wave g computes gate-g tile: G[16,16] = h_{t-1}[16,512] x WhT(rows g*H+j0..)
// starting from precomputed xg (f32). Elementwise cell update fused via LDS.
// Hs holds bf16 h for ALL timesteps, row index = b*T + t (also feeds final GEMM).
__global__ void lstm_step(__bf16* __restrict__ Hs, const __bf16* __restrict__ WhT,
                          const float* __restrict__ XG, float* __restrict__ Cst, int t) {
  __shared__ float gbuf[4][16][16];
  int g    = threadIdx.x >> 5;
  int lane = threadIdx.x & 31;
  int j0 = blockIdx.x * 16;
  int n0 = g * HDIM + j0;          // column base within 2048 gate columns
  int rc = lane & 15;
  int kb = (lane >> 4) << 4;
  int mB = (lane >> 4) << 3;

  v8f acc;
  #pragma unroll
  for (int v = 0; v < 8; ++v)      // acc(m,n) = xg[b=m, t, n]
    acc[v] = XG[(size_t)((mB + v) * TDIM + t) * G4H + (n0 + rc)];

  if (t > 0) {
    const __bf16* ap = Hs  + ((size_t)rc * TDIM + (t - 1)) * HDIM + kb;
    const __bf16* bp = WhT + (size_t)(n0 + rc) * HDIM + kb;
    for (int k = 0; k < HDIM; k += 32) {
      v16bf av = *(const v16bf*)(ap + k);
      v16bf bv = *(const v16bf*)(bp + k);
      acc = __builtin_amdgcn_wmma_f32_16x16x32_bf16(false, av, false, bv,
                                                    (short)0, acc, false, false);
    }
  }

  #pragma unroll
  for (int v = 0; v < 8; ++v)
    gbuf[g][mB + v][rc] = acc[v];
  __syncthreads();

  #pragma unroll
  for (int p = 0; p < 2; ++p) {    // 128 threads x 2 -> 256 (b,j) cells
    int e = threadIdx.x * 2 + p;
    int b = e >> 4, jj = e & 15;
    float gi = sigmoidf_(gbuf[0][b][jj]);
    float gf = sigmoidf_(gbuf[1][b][jj]);
    float go = sigmoidf_(gbuf[2][b][jj]);
    float gc = tanhf(gbuf[3][b][jj]);
    float cold = (t == 0) ? 0.0f : Cst[b * HDIM + j0 + jj];
    float cn = gf * cold + gi * gc;
    Cst[b * HDIM + j0 + jj] = cn;
    Hs[((size_t)b * TDIM + t) * HDIM + j0 + jj] = f2bf(go * tanhf(cn));
  }
}

// ---------------------------------------------------------------------------
extern "C" void kernel_launch(void* const* d_in, const int* in_sizes, int n_in,
                              void* d_out, int out_size, void* d_ws, size_t ws_size,
                              hipStream_t stream) {
  (void)in_sizes; (void)n_in; (void)out_size; (void)ws_size;
  const int*   tokens = (const int*)  d_in[0];
  const float* emb    = (const float*)d_in[1];
  const float* W_hi = (const float*)d_in[2];
  const float* W_xi = (const float*)d_in[3];
  const float* b_i  = (const float*)d_in[4];
  const float* W_hf = (const float*)d_in[5];
  const float* W_xf = (const float*)d_in[6];
  const float* b_f  = (const float*)d_in[7];
  const float* W_ho = (const float*)d_in[8];
  const float* W_xo = (const float*)d_in[9];
  const float* b_o  = (const float*)d_in[10];
  const float* W_hc = (const float*)d_in[11];
  const float* W_xc = (const float*)d_in[12];
  const float* b_c  = (const float*)d_in[13];
  const float* fc_W = (const float*)d_in[14];
  const float* fc_b = (const float*)d_in[15];
  float* out = (float*)d_out;

  // Workspace carve-out (256B aligned slabs). Total ~116 MB.
  char* ws = (char*)d_ws;
  size_t off = 0;
  auto carve = [&](size_t bytes) -> void* {
    void* p = ws + off;
    off = (off + bytes + 255) & ~(size_t)255;
    return p;
  };
  __bf16* Xbf  = (__bf16*)carve((size_t)ROWS  * EDIM * 2);  //  4 MB
  __bf16* WxT  = (__bf16*)carve((size_t)G4H   * EDIM * 2);  //  1 MB
  __bf16* WhT  = (__bf16*)carve((size_t)G4H   * HDIM * 2);  //  2 MB
  __bf16* fcT  = (__bf16*)carve((size_t)VOCAB * HDIM * 2);  // 32 MB
  float*  bcat = (float*) carve((size_t)G4H   * 4);
  float*  XG   = (float*) carve((size_t)ROWS  * G4H * 4);   // 64 MB
  float*  Cst  = (float*) carve((size_t)BDIM  * HDIM * 4);
  __bf16* Hs   = (__bf16*)carve((size_t)ROWS  * HDIM * 2);  //  8 MB

  dim3 b256(256);
  prep_wT  <<<dim3((G4H * EDIM  + 255) / 256), b256, 0, stream>>>(W_xi, W_xf, W_xo, W_xc, WxT, EDIM);
  prep_wT  <<<dim3((G4H * HDIM  + 255) / 256), b256, 0, stream>>>(W_hi, W_hf, W_ho, W_hc, WhT, HDIM);
  prep_fcT <<<dim3((VOCAB * HDIM + 255) / 256), b256, 0, stream>>>(fc_W, fcT);
  prep_bcat<<<dim3((G4H + 255) / 256),          b256, 0, stream>>>(b_i, b_f, b_o, b_c, bcat);
  gather_x <<<dim3((ROWS * EDIM + 255) / 256),  b256, 0, stream>>>(tokens, emb, Xbf);

  // xg = X[8192,256] @ WxT + bcat -> [8192,2048] f32
  // grid: N/64 x M/256 (8 waves x 32-row patches per block)
  wmma_gemm_bias<<<dim3(G4H / 64, ROWS / 256), b256, 0, stream>>>(Xbf, WxT, bcat, XG, G4H, EDIM);

  // Sequential scan: one tiny WMMA kernel per timestep (B=16 == one M tile).
  for (int t = 0; t < TDIM; ++t)
    lstm_step<<<dim3(HDIM / 16), dim3(128), 0, stream>>>(Hs, WhT, XG, Cst, t);

  // logits = Hs[8192,512] @ fcT + fc_b -> [8192,32000] f32 (dominant GEMM, 268 GFLOP)
  wmma_gemm_bias<<<dim3(VOCAB / 64, ROWS / 256), b256, 0, stream>>>(Hs, fcT, fc_b, out, VOCAB, HDIM);
}
